// TT_33818572488912
// MI455X (gfx1250) — matrix operand order repaired and checked by reading, not compiled, not run on hardware
//
#include <hip/hip_runtime.h>
#include <hip/hip_bf16.h>

typedef __attribute__((ext_vector_type(16))) _Float16 v16h;
typedef __attribute__((ext_vector_type(8)))  float    v8f;

#define BATCHES 512
#define NIN     64      // INPUT_SIZE
#define AD      64      // ALPHA_DIM
#define STEPS   63      // chain matrices (positions 0..62)

#define INV_LN2 1.44269504088896340736f
#define LN2     0.69314718055994530942f

#if defined(__gfx1250__) && __has_builtin(__builtin_amdgcn_global_load_async_to_lds_b128)
#define USE_ASYNC_LDS 1
typedef int v4i_ __attribute__((vector_size(16)));
typedef __attribute__((address_space(1))) v4i_* gv4p;   // global int4*
typedef __attribute__((address_space(3))) v4i_* lv4p;   // LDS int4*
#else
#define USE_ASYNC_LDS 0
#endif

__device__ __forceinline__ float fast_exp2(float x) {
#if __has_builtin(__builtin_amdgcn_exp2f)
    return __builtin_amdgcn_exp2f(x);      // raw v_exp_f32
#else
    return exp2f(x);
#endif
}
__device__ __forceinline__ float fast_log2(float x) {
#if __has_builtin(__builtin_amdgcn_logf)
    return __builtin_amdgcn_logf(x);       // raw v_log_f32 (log2)
#else
    return log2f(x);
#endif
}

#if USE_ASYNC_LDS
__device__ __forceinline__ void wait_async_le2() {
#if __has_builtin(__builtin_amdgcn_s_wait_asynccnt)
    __builtin_amdgcn_s_wait_asynccnt(2);
#else
    asm volatile("s_wait_asynccnt 0x2" ::: "memory");
#endif
}
__device__ __forceinline__ void wait_async_le0() {
#if __has_builtin(__builtin_amdgcn_s_wait_asynccnt)
    __builtin_amdgcn_s_wait_asynccnt(0);
#else
    asm volatile("s_wait_asynccnt 0x0" ::: "memory");
#endif
}
#endif

__device__ __forceinline__ float logpx(float t, float xv) {
    // -clip(t,0) + t*x - log1p(exp(-|t|))   (nats)
    return -fmaxf(t, 0.0f) + t * xv - log1pf(__expf(-fabsf(t)));
}

// ---------------------------------------------------------------------------
// Kernel 1: one workgroup per (step n, x-bit). Computes, in LOG2 domain,
//   M2[i][j] = (logpx(theta[n,i,j],x) + log_softmax(u[n])[j]) / ln2
// then colmax[j] = max_i M2[i][j], Bexp[col][k] = 2^(M2[k][col]-colmax[col])
// stored COLUMN-major f16 (contiguous WMMA B-fragment loads).
// n==0 also stores the log2-domain seed matrices.
// ---------------------------------------------------------------------------
__global__ __launch_bounds__(256)
void precompute_kernel(const float* __restrict__ theta,
                       const float* __restrict__ u,       // (63,64)
                       _Float16*    __restrict__ Bexp,    // [126][64][64] col-major
                       float*       __restrict__ colmax,  // [126][64]  (log2)
                       float*       __restrict__ M0log)   // [2][64][64] (log2)
{
    const int n  = blockIdx.x >> 1;   // 0..62
    const int xv = blockIdx.x & 1;
    const int t  = threadIdx.x;

    __shared__ float Ml[AD][AD];
    __shared__ float ub[AD];
    __shared__ float cmx[AD];

    if (t < AD) ub[t] = u[n * AD + t];
    __syncthreads();

    // log-softmax denominator (redundant per thread; broadcast LDS reads)
    float m = -__builtin_inff();
    for (int j = 0; j < AD; ++j) m = fmaxf(m, ub[j]);
    float s = 0.0f;
    for (int j = 0; j < AD; ++j) s += __expf(ub[j] - m);
    const float lse = m + __logf(s);

    const int r  = t >> 2;
    const int c0 = (t & 3) * 16;
    const float xf = (float)xv;
    #pragma unroll 4
    for (int c = c0; c < c0 + 16; ++c) {
        float th = theta[n * AD * AD + r * AD + c];
        float v  = (logpx(th, xf) + (ub[c] - lse)) * INV_LN2;   // log2 domain
        Ml[r][c] = v;
        if (n == 0) M0log[(xv * AD + r) * AD + c] = v;
    }
    __syncthreads();

    if (t < AD) {                      // column max (conflict-free column scan)
        float cm = -__builtin_inff();
        for (int i = 0; i < AD; ++i) cm = fmaxf(cm, Ml[i][t]);
        cmx[t] = cm;
        colmax[blockIdx.x * AD + t] = cm;
    }
    __syncthreads();

    // Bexp column-major: dst[col][k] = 2^(Ml[k][col] - cmx[col])  (f16)
    const int col = t >> 2;
    const int k0  = (t & 3) * 16;
    _Float16* dst = Bexp + (size_t)blockIdx.x * AD * AD + col * AD + k0;
    const float cm = cmx[col];
    #pragma unroll 4
    for (int k = 0; k < 16; ++k)
        dst[k] = (_Float16)fast_exp2(Ml[k0 + k][col] - cm);
}

// ---------------------------------------------------------------------------
// Kernel 2: one workgroup (8 wave32) per batch element. Running 64x64 "left"
// (log2 domain) in LDS; 62 chained log-semiring matmuls via f16 WMMA.
// B tiles double-buffered in LDS via GLOBAL_LOAD_ASYNC_TO_LDS_B128 prefetch.
// ---------------------------------------------------------------------------
__global__ __launch_bounds__(256)
void chain_kernel(const float*    __restrict__ x,
                  const float*    __restrict__ theta,
                  const _Float16* __restrict__ Bexp,
                  const float*    __restrict__ colmax,
                  const float*    __restrict__ M0log,
                  float*          __restrict__ out)
{
    const int b    = blockIdx.x;
    const int t    = threadIdx.x;
    const int lane = t & 31;
    const int wave = t >> 5;

    __shared__ float                 left[AD][AD];     // 16 KB (log2 domain)
    __shared__ __align__(16) _Float16 Ae[AD][AD];      //  8 KB 2^(left-rowmax)
    __shared__ __align__(16) _Float16 Bl[2][AD][AD];   // 16 KB, [buf][col][k]
    __shared__ float part[AD][4];
    __shared__ float rmax[AD];
    __shared__ float cmx[2][AD];
    __shared__ float vb[AD];
    __shared__ int   xbits[NIN];

    const int r  = t >> 2;
    const int q  = t & 3;
    const int c0 = q * 16;

    if (t < NIN) xbits[t] = (x[b * NIN + t] > 0.5f) ? 1 : 0;
    __syncthreads();

    // seed: left = log2-domain matrix for position 0, selected by x[b,0]
    {
        const float* src = M0log + xbits[0] * AD * AD + r * AD + c0;
        #pragma unroll
        for (int c = 0; c < 16; ++c) left[r][c0 + c] = src[c];
    }

#if USE_ASYNC_LDS
    // prologue: async-stage B(mat for n=1) into buffer 0 (each thread 32 B)
    {
        const int mat1 = 2 + xbits[1];
        const _Float16* g = Bexp + (size_t)mat1 * AD * AD + t * 16;
        _Float16*       l = &Bl[0][0][0] + t * 16;
        __builtin_amdgcn_global_load_async_to_lds_b128((gv4p)g, (lv4p)l, 0, 0);
        __builtin_amdgcn_global_load_async_to_lds_b128((gv4p)g, (lv4p)l, 16, 0);
        if (t < AD) cmx[0][t] = colmax[mat1 * AD + t];
    }
#endif
    __syncthreads();

    const int tile0 = wave * 2;          // 16 output tiles, 2 per wave

    for (int n = 1; n < STEPS; ++n) {
        const int buf = (n - 1) & 1;
#if USE_ASYNC_LDS
        // prefetch NEXT step's B tile + colmax (x-bit known in advance)
        float cmxreg = 0.0f;
        if (n + 1 < STEPS) {
            const int nmat = 2 * (n + 1) + xbits[n + 1];
            const _Float16* g = Bexp + (size_t)nmat * AD * AD + t * 16;
            _Float16*       l = &Bl[buf ^ 1][0][0] + t * 16;
            __builtin_amdgcn_global_load_async_to_lds_b128((gv4p)g, (lv4p)l, 0, 0);
            __builtin_amdgcn_global_load_async_to_lds_b128((gv4p)g, (lv4p)l, 16, 0);
            if (t < AD) cmxreg = colmax[nmat * AD + t];
        }
#else
        // synchronous staging of the current tile
        const int mat = 2 * n + xbits[n];
        {
            const uint4* src = (const uint4*)(Bexp + (size_t)mat * AD * AD);
            uint4* dst = (uint4*)(&Bl[buf][0][0]);
            dst[t * 2 + 0] = src[t * 2 + 0];
            dst[t * 2 + 1] = src[t * 2 + 1];
        }
        if (t < AD) cmx[buf][t] = colmax[mat * AD + t];
#endif
        // row max: partial over 16-col strip, combine after barrier
        float pm = -__builtin_inff();
        #pragma unroll
        for (int c = 0; c < 16; ++c) pm = fmaxf(pm, left[r][c0 + c]);
        part[r][q] = pm;
        __syncthreads();

        const float rm = fmaxf(fmaxf(part[r][0], part[r][1]),
                               fmaxf(part[r][2], part[r][3]));
        if (q == 0) rmax[r] = rm;
        #pragma unroll
        for (int c = 0; c < 16; ++c)
            Ae[r][c0 + c] = (_Float16)fast_exp2(left[r][c0 + c] - rm);

#if USE_ASYNC_LDS
        if (n + 1 < STEPS) {
            if (t < AD) cmx[buf ^ 1][t] = cmxreg;
            wait_async_le2();            // current buffer done; next still in flight
        } else {
            wait_async_le0();
        }
#endif
        __syncthreads();

        // WMMA stage: C(16x16,f32) = Ae(16x64) * Bl[buf](64x16), 2 tiles/wave
        #pragma unroll
        for (int tt = 0; tt < 2; ++tt) {
            const int tile = tile0 + tt;
            const int tr = (tile >> 2) * 16;
            const int tc = (tile & 3) * 16;
            v8f acc = {};
            #pragma unroll
            for (int kc = 0; kc < 2; ++kc) {
                const int k0 = kc * 32;
                // A fragment (ISA 16-bit A 16x32 layout)
                const int arow = tr + (lane & 15);
                const int klo  = k0 + ((lane >> 4) & 1) * 8;
                v16h a, bf;
                {
                    union { uint4 u[2]; v16h h; } cv;
                    cv.u[0] = *(const uint4*)&Ae[arow][klo];
                    cv.u[1] = *(const uint4*)&Ae[arow][klo + 16];
                    a = cv.h;
                }
                // B fragment: lane%16 = column, lane-half selects K 0-7,16-23 / 8-15,24-31
                const int bcol = tc + (lane & 15);
                const int ks   = k0 + ((lane >> 4) & 1) * 8;
                {
                    union { uint4 u[2]; v16h h; } cv;
                    cv.u[0] = *(const uint4*)&Bl[buf][bcol][ks];
                    cv.u[1] = *(const uint4*)&Bl[buf][bcol][ks + 16];
                    bf = cv.h;
                }
                acc = __builtin_amdgcn_wmma_f32_16x16x32_f16(
                        false, a, false, bf, (short)0, acc, false, false);
            }
            // write back: left = log2(acc) + rowmax + colmax  (C layout)
            const int crow0 = tr + ((lane >> 4) & 1) * 8;
            const int ccol  = tc + (lane & 15);
            #pragma unroll
            for (int rr = 0; rr < 8; ++rr)
                left[crow0 + rr][ccol] =
                    fast_log2(acc[rr]) + rmax[crow0 + rr] + cmx[buf][ccol];
        }
        __syncthreads();
    }

    // final contraction: out[b][i] = ln2 * log2sumexp2_k(left[i][k] + v2[k]),
    // v2[k] = logpx(theta[63,k,0], x[b,63]) / ln2
    if (t < AD) {
        const float xf = (float)xbits[NIN - 1];
        vb[t] = logpx(theta[(NIN - 1) * AD * AD + t * AD + 0], xf) * INV_LN2;
    }
    __syncthreads();
    if (t < AD) {
        float m = -__builtin_inff();
        for (int k = 0; k < AD; ++k) m = fmaxf(m, left[t][k] + vb[k]);
        float s = 0.0f;
        for (int k = 0; k < AD; ++k) s += fast_exp2(left[t][k] + vb[k] - m);
        out[b * AD + t] = LN2 * (m + fast_log2(s));
    }
}

// ---------------------------------------------------------------------------
extern "C" void kernel_launch(void* const* d_in, const int* in_sizes, int n_in,
                              void* d_out, int out_size, void* d_ws, size_t ws_size,
                              hipStream_t stream)
{
    const float* x     = (const float*)d_in[0];   // (512,64)
    const float* theta = (const float*)d_in[1];   // (1,64,64,64)
    const float* u     = (const float*)d_in[2];   // (1,63,1,64)

    char* ws = (char*)d_ws;
    _Float16* Bexp   = (_Float16*)ws;                       // 126*4096*2  = 1032192 B
    float*    colmax = (float*)(ws + 1032192);              // 126*64*4    =   32256 B
    float*    M0log  = (float*)(ws + 1032192 + 32256);      // 2*4096*4    =   32768 B
    float*    out    = (float*)d_out;                       // (512,64)

    precompute_kernel<<<dim3(STEPS * 2), dim3(256), 0, stream>>>(theta, u, Bexp, colmax, M0log);
    chain_kernel<<<dim3(BATCHES), dim3(256), 0, stream>>>(x, theta, Bexp, colmax, M0log, out);
}
